// Attention_30588757082271
// MI455X (gfx1250) — matrix operand order repaired and checked by reading, not compiled, not run on hardware
//
#include <hip/hip_runtime.h>

// ---------------------------------------------------------------------------
// MI455X (gfx1250) attention block:  x@Wq/Wk/Wv -> RoPE -> causal GQA flash
// attention -> @Wo.  All GEMM work on v_wmma_f32_16x16x32_f16 (f32 accum).
// V tiles staged to LDS with GLOBAL_LOAD_ASYNC_TO_LDS_B128 (ASYNCcnt).
// ---------------------------------------------------------------------------

typedef __attribute__((ext_vector_type(16))) _Float16 v16h;
typedef __attribute__((ext_vector_type(8)))  _Float16 v8h;
typedef __attribute__((ext_vector_type(8)))  float    v8f;

static constexpr int Bc   = 2;
static constexpr int Sc   = 2048;
static constexpr int Dc   = 2048;
static constexpr int Hc   = 16;
static constexpr int KVHc = 8;
static constexpr int HDc  = 128;
static constexpr int NREP = Hc / KVHc;           // 2
static constexpr float SCALE   = 0.08838834764831845f;  // 1/sqrt(HD)
static constexpr float NEG_INF = -1000000000.0f;

union U16h { v16h v; v8h h8[2]; _Float16 h[16]; };
union U8f  { v8f  v; float f[8]; };

// ---------------------------------------------------------------------------
// fp32 -> f16 elementwise convert
// ---------------------------------------------------------------------------
__global__ void cvt_f32_f16(const float* __restrict__ src,
                            _Float16* __restrict__ dst, int n) {
  int i = blockIdx.x * blockDim.x + threadIdx.x;
  if (i < n) dst[i] = (_Float16)src[i];
}

// fp32 W[K][N] -> f16 WT[N][K]   (so WMMA B-frags become one contiguous load)
__global__ void transpose_cvt_f32_f16(const float* __restrict__ w,
                                      _Float16* __restrict__ wt,
                                      int K, int N) {
  int i = blockIdx.x * blockDim.x + threadIdx.x;
  if (i >= K * N) return;
  int n = i % N;
  int k = i / N;
  wt[(size_t)n * K + k] = (_Float16)w[i];
}

// ---------------------------------------------------------------------------
// WMMA GEMM:  C[M][N] = A[M][K] * BT[N][K]^T   (f16 in, f32 accum)
// One wave per 16x64 C strip. All frags for a k-step are preloaded before the
// WMMA chain so the scheduler can clause the loads and overlap with the
// matrix pipe.  grid = (N/64, M/16), block = 32.
// ---------------------------------------------------------------------------
template <typename OutT>
__global__ __launch_bounds__(32, 1) void wmma_gemm(
    const _Float16* __restrict__ A,
    const _Float16* __restrict__ BT,
    OutT* __restrict__ C,
    int M, int N, int K) {
  const int lane = threadIdx.x;
  const int l15  = lane & 15;
  const int hi   = lane >> 4;
  const int tn0  = blockIdx.x * 4;      // first of 4 16-wide N tiles
  const int tm   = blockIdx.y;

  const _Float16* arow  = A + (size_t)(tm * 16 + l15) * K;
  const _Float16* brow0 = BT + (size_t)(tn0 * 16 + l15) * K;
  const size_t bstride = (size_t)16 * K;   // next N tile = +16 rows of BT

  v8f acc0 = {}, acc1 = {}, acc2 = {}, acc3 = {};
  for (int k0 = 0; k0 < K; k0 += 32) {
    if (k0 + 256 < K) {
      __builtin_prefetch(arow + k0 + 256, 0, 1);   // global_prefetch_b8
      __builtin_prefetch(brow0 + k0 + 256, 0, 1);
    }
    // ---- preload all frags for this k-step ----
    U16h a, b0, b1, b2, b3;
    a.h8[0] = *(const v8h*)(arow + k0 + 16 * hi);
    a.h8[1] = *(const v8h*)(arow + k0 + 16 * hi + 8);
    b0.v = *(const v16h*)(brow0 + k0 + 16 * hi);
    b1.v = *(const v16h*)(brow0 + bstride + k0 + 16 * hi);
    b2.v = *(const v16h*)(brow0 + 2 * bstride + k0 + 16 * hi);
    b3.v = *(const v16h*)(brow0 + 3 * bstride + k0 + 16 * hi);
    // ---- matrix pipe ----
    acc0 = __builtin_amdgcn_wmma_f32_16x16x32_f16(
        false, a.v, false, b0.v, (short)0, acc0, false, false);
    acc1 = __builtin_amdgcn_wmma_f32_16x16x32_f16(
        false, a.v, false, b1.v, (short)0, acc1, false, false);
    acc2 = __builtin_amdgcn_wmma_f32_16x16x32_f16(
        false, a.v, false, b2.v, (short)0, acc2, false, false);
    acc3 = __builtin_amdgcn_wmma_f32_16x16x32_f16(
        false, a.v, false, b3.v, (short)0, acc3, false, false);
  }
  // C layout: lane holds col n = l15, rows r + 8*hi in element r
  U8f res[4];
  res[0].v = acc0; res[1].v = acc1; res[2].v = acc2; res[3].v = acc3;
  for (int t = 0; t < 4; ++t)
    for (int r = 0; r < 8; ++r) {
      int m = tm * 16 + r + 8 * hi;
      int n = (tn0 + t) * 16 + l15;
      C[(size_t)m * N + n] = (OutT)res[t].f[r];
    }
}

// ---------------------------------------------------------------------------
// RoPE (reference's exact formula):
//   even' = e*sin - o*cos ;  odd' = e*cos + o*sin
// t layout: [B][S][nheads][HD] f16 in-place; fc: [S][HD/2][2] f32 (cos,sin)
// ---------------------------------------------------------------------------
__global__ void rope_kernel(_Float16* __restrict__ t,
                            const float* __restrict__ fc,
                            int nheads, int total) {
  int idx = blockIdx.x * blockDim.x + threadIdx.x;
  if (idx >= total) return;
  int p    = idx & 63;                 // HD/2 = 64 pairs
  int rest = idx >> 6;                 // (b*S + s)*nheads + head
  int s    = (rest / nheads) % Sc;
  size_t base = (size_t)idx * 2;
  float e = (float)t[base];
  float o = (float)t[base + 1];
  float c  = fc[(s * 64 + p) * 2 + 0];
  float si = fc[(s * 64 + p) * 2 + 1];
  t[base]     = (_Float16)(e * si - o * c);
  t[base + 1] = (_Float16)(e * c + o * si);
}

// ---------------------------------------------------------------------------
// Causal GQA flash attention. One wave per (b, h, 16-query tile).
// qh:[B][S][H][HD]  kh,vh:[B][S][KVH][HD]  oh:[B][S][H][HD]  (all f16)
// ---------------------------------------------------------------------------
__global__ __launch_bounds__(32, 1) void flash_attn(
    const _Float16* __restrict__ qh,
    const _Float16* __restrict__ kh,
    const _Float16* __restrict__ vh,
    _Float16* __restrict__ oh) {
  const int lane = threadIdx.x;
  const int l15  = lane & 15;
  const int hi   = lane >> 4;
  const int qt   = blockIdx.x;     // S/16 tiles
  const int h    = blockIdx.y;
  const int b    = blockIdx.z;
  const int kvh  = h / NREP;

  __shared__ _Float16 pt[16 * 32];      // probs tile (C-layout -> A-layout)
  __shared__ _Float16 vt[32 * HDc];     // V tile: 32 keys x 128

  // ---- load Q A-frags once (4 chunks of K=32 over HD=128) ----
  U16h aQ[4];
  {
    const _Float16* qrow =
        qh + (((size_t)(b * Sc + qt * 16 + l15)) * Hc + h) * HDc;
    for (int c = 0; c < 4; ++c) {
      aQ[c].h8[0] = *(const v8h*)(qrow + c * 32 + 16 * hi);
      aQ[c].h8[1] = *(const v8h*)(qrow + c * 32 + 16 * hi + 8);
    }
  }

  U8f o[8];
  for (int t = 0; t < 8; ++t) o[t].v = (v8f){};
  float mrun[8], lrun[8];
  for (int r = 0; r < 8; ++r) { mrun[r] = -1e30f; lrun[r] = 0.f; }

  const int lastkey = qt * 16 + 15;
  const int nchunks = (lastkey + 32) / 32;      // 32 keys per chunk

  for (int jc = 0; jc < nchunks; ++jc) {
    const int j0 = jc * 32;

    // ---- async-copy V tile (32 keys x 128) to LDS: lane owns one key row,
    //      copied as 16 x b128 async transfers tracked by ASYNCcnt ----
    {
      const _Float16* vrow =
          vh + (((size_t)(b * Sc + j0 + lane)) * KVHc + kvh) * HDc;
      unsigned lds_off = (unsigned)(uintptr_t)&vt[lane * HDc];
      unsigned long long ga = (unsigned long long)(uintptr_t)vrow;
      for (int i = 0; i < 16; ++i) {
        asm volatile("global_load_async_to_lds_b128 %0, %1, off"
                     :: "v"(lds_off), "v"(ga) : "memory");
        lds_off += 16;
        ga += 16;
      }
    }

    // ---- scores S = Q K^T : preload all 8 K B-frags, then 8 WMMAs ----
    const _Float16* kb0 =
        kh + (((size_t)(b * Sc + j0 + l15)) * KVHc + kvh) * HDc;
    const _Float16* kb1 =
        kh + (((size_t)(b * Sc + j0 + 16 + l15)) * KVHc + kvh) * HDc;
    U16h bk0[4], bk1[4];
    for (int c = 0; c < 4; ++c) {
      bk0[c].v = *(const v16h*)(kb0 + c * 32 + 16 * hi);
      bk1[c].v = *(const v16h*)(kb1 + c * 32 + 16 * hi);
    }
    v8f s0v = {}, s1v = {};
    for (int c = 0; c < 4; ++c) {
      s0v = __builtin_amdgcn_wmma_f32_16x16x32_f16(
          false, aQ[c].v, false, bk0[c].v, (short)0, s0v, false, false);
      s1v = __builtin_amdgcn_wmma_f32_16x16x32_f16(
          false, aQ[c].v, false, bk1[c].v, (short)0, s1v, false, false);
    }
    U8f s0, s1; s0.v = s0v; s1.v = s1v;

    // ---- mask + scale + online softmax (rows r+8*hi, col l15) ----
    float p0[8], p1[8], alpha[8];
    for (int r = 0; r < 8; ++r) {
      int mrow = qt * 16 + r + 8 * hi;
      float v0 = s0.f[r] * SCALE + ((j0 + l15)      > mrow ? NEG_INF : 0.f);
      float v1 = s1.f[r] * SCALE + ((j0 + 16 + l15) > mrow ? NEG_INF : 0.f);
      float tmax = fmaxf(v0, v1);
      for (int m = 1; m <= 8; m <<= 1)
        tmax = fmaxf(tmax, __shfl_xor(tmax, m, 32));   // 16-lane groups
      float mnew = fmaxf(mrun[r], tmax);
      alpha[r] = __expf(mrun[r] - mnew);
      p0[r] = __expf(v0 - mnew);
      p1[r] = __expf(v1 - mnew);
      float rs = p0[r] + p1[r];
      for (int m = 1; m <= 8; m <<= 1)
        rs += __shfl_xor(rs, m, 32);
      lrun[r] = lrun[r] * alpha[r] + rs;
      mrun[r] = mnew;
    }
    for (int t = 0; t < 8; ++t)
      for (int r = 0; r < 8; ++r) o[t].f[r] *= alpha[r];

    // ---- P: C-layout -> A-layout through LDS ----
    for (int r = 0; r < 8; ++r) {
      int row = r + 8 * hi;
      pt[row * 32 + l15]      = (_Float16)p0[r];
      pt[row * 32 + 16 + l15] = (_Float16)p1[r];
    }
    // P stores (DScnt) and async V tile (ASYNCcnt) must land before gathers
    asm volatile("s_wait_dscnt 0x0" ::: "memory");
    asm volatile("s_wait_asynccnt 0x0" ::: "memory");

    U16h aP;
    aP.h8[0] = *(const v8h*)&pt[l15 * 32 + 16 * hi];       // keys 16*hi+0..7
    aP.h8[1] = *(const v8h*)&pt[l15 * 32 + 16 * hi + 8];   // keys 16*hi+8..15

    // ---- O += P V : 8 output tiles across HD=128 ----
    for (int nt = 0; nt < 8; ++nt) {
      U16h bv;
      int hd = nt * 16 + l15;
      for (int i = 0; i < 16; ++i)
        bv.h[i] = vt[(16 * hi + i) * HDc + hd];            // key = 16*hi+i
      o[nt].v = __builtin_amdgcn_wmma_f32_16x16x32_f16(
          false, aP.v, false, bv.v, (short)0, o[nt].v, false, false);
    }
  }

  // ---- normalize and store ----
  for (int nt = 0; nt < 8; ++nt)
    for (int r = 0; r < 8; ++r) {
      int srow = qt * 16 + r + 8 * hi;
      float val = o[nt].f[r] / lrun[r];
      oh[(((size_t)(b * Sc + srow)) * Hc + h) * HDc + nt * 16 + l15] =
          (_Float16)val;
    }
}

// ---------------------------------------------------------------------------
// Host orchestration
// ---------------------------------------------------------------------------
extern "C" void kernel_launch(void* const* d_in, const int* in_sizes, int n_in,
                              void* d_out, int out_size, void* d_ws,
                              size_t ws_size, hipStream_t stream) {
  const float* x  = (const float*)d_in[0];
  const float* fc = (const float*)d_in[1];
  // d_in[2] = mask (causal, reproduced analytically)
  const float* wq = (const float*)d_in[3];
  const float* wk = (const float*)d_in[4];
  const float* wv = (const float*)d_in[5];
  const float* wo = (const float*)d_in[6];
  float* out = (float*)d_out;

  const int M  = Bc * Sc;          // 4096 rows
  const int NQ = Hc * HDc;         // 2048
  const int NK = KVHc * HDc;       // 1024

  // workspace layout (halves); oh aliases xh (x only needed for QKV GEMMs)
  _Float16* ws  = (_Float16*)d_ws;
  _Float16* xh  = ws;                                    size_t off = (size_t)M * Dc;
  _Float16* oh  = xh;                                    // alias, used later
  _Float16* wqT = ws + off;  off += (size_t)NQ * Dc;
  _Float16* wkT = ws + off;  off += (size_t)NK * Dc;
  _Float16* wvT = ws + off;  off += (size_t)NK * Dc;
  _Float16* woT = ws + off;  off += (size_t)Dc * NQ;
  _Float16* qh  = ws + off;  off += (size_t)M * NQ;
  _Float16* kh  = ws + off;  off += (size_t)M * NK;
  _Float16* vh  = ws + off;  off += (size_t)M * NK;
  (void)ws_size; (void)n_in; (void)in_sizes; (void)out_size;

  const int T = 256;
  // 1. convert x and transpose-convert weights
  {
    int n = M * Dc;
    cvt_f32_f16<<<(n + T - 1) / T, T, 0, stream>>>(x, xh, n);
    n = Dc * NQ;
    transpose_cvt_f32_f16<<<(n + T - 1) / T, T, 0, stream>>>(wq, wqT, Dc, NQ);
    n = Dc * NK;
    transpose_cvt_f32_f16<<<(n + T - 1) / T, T, 0, stream>>>(wk, wkT, Dc, NK);
    transpose_cvt_f32_f16<<<(n + T - 1) / T, T, 0, stream>>>(wv, wvT, Dc, NK);
    n = NQ * Dc;
    transpose_cvt_f32_f16<<<(n + T - 1) / T, T, 0, stream>>>(wo, woT, NQ, Dc);
  }

  // 2. QKV projections (WMMA, 16x64 strip per wave)
  wmma_gemm<_Float16><<<dim3(NQ / 64, M / 16), 32, 0, stream>>>(
      xh, wqT, qh, M, NQ, Dc);
  wmma_gemm<_Float16><<<dim3(NK / 64, M / 16), 32, 0, stream>>>(
      xh, wkT, kh, M, NK, Dc);
  wmma_gemm<_Float16><<<dim3(NK / 64, M / 16), 32, 0, stream>>>(
      xh, wvT, vh, M, NK, Dc);

  // 3. RoPE on Q and K
  {
    int nq = Bc * Sc * Hc * (HDc / 2);
    rope_kernel<<<(nq + T - 1) / T, T, 0, stream>>>(qh, fc, Hc, nq);
    int nk = Bc * Sc * KVHc * (HDc / 2);
    rope_kernel<<<(nk + T - 1) / T, T, 0, stream>>>(kh, fc, KVHc, nk);
  }

  // 4. flash attention (overwrites xh region as oh; x no longer needed)
  flash_attn<<<dim3(Sc / 16, Hc, Bc), 32, 0, stream>>>(qh, kh, vh, oh);

  // 5. output projection -> fp32
  wmma_gemm<float><<<dim3(Dc / 16 / 4, M / 16), 32, 0, stream>>>(
      oh, woT, out, M, Dc, NQ);
}
